// ArcticMoE_2113123910131
// MI455X (gfx1250) — matrix operand order repaired and checked by reading, not compiled, not run on hardware
//
#include <hip/hip_runtime.h>

typedef __attribute__((ext_vector_type(16))) __bf16 v16bf;
typedef __attribute__((ext_vector_type(8)))  __bf16 v8bf;
typedef __attribute__((ext_vector_type(4)))  __bf16 v4bf;
typedef __attribute__((ext_vector_type(8)))  float  v8f;
typedef __attribute__((ext_vector_type(4)))  float  v4f;

#define T_TOK 4096
#define H_DIM 2048
#define F_DIM 4096
#define E_NUM 8

#define BM 128
#define BN 64
#define BK 32
#define LDSA 40   // bf16 elements per row (32 + 8 pad)
#define LDSB 40

union Frag { v16bf v; v8bf h[2]; };

// Async copy 16B global -> LDS (tracked by ASYNCcnt). INST_OFFSET applies to
// both the global and LDS address per the ISA pseudocode.
#define ASYNC_CP16(ldsoff, gaddr)                                              \
  asm volatile("global_load_async_to_lds_b128 %0, %1, off"                     \
               :: "v"(ldsoff), "v"(gaddr) : "memory")
#define ASYNC_CP16_OFF16(ldsoff, gaddr)                                        \
  asm volatile("global_load_async_to_lds_b128 %0, %1, off offset:16"           \
               :: "v"(ldsoff), "v"(gaddr) : "memory")
#define WAIT_ASYNC0() asm volatile("s_wait_asynccnt 0x0" ::: "memory")

// ---------------------------------------------------------------- f32 -> bf16
__global__ __launch_bounds__(256) void arctic_cvt_bf16(
    const float* __restrict__ in, __bf16* __restrict__ out, int n) {
  int i = (blockIdx.x * 256 + threadIdx.x) * 4;
  if (i < n) {
    v4f v = *(const v4f*)(in + i);
    v4bf o;
    o[0] = (__bf16)v[0]; o[1] = (__bf16)v[1];
    o[2] = (__bf16)v[2]; o[3] = (__bf16)v[3];
    *(v4bf*)(out + i) = o;
  }
}

// ---------------------------------------------------------------- router
__global__ __launch_bounds__(256) void arctic_router(
    const float* __restrict__ x, const float* __restrict__ gw,
    float* __restrict__ probs, float* __restrict__ padw,
    int* __restrict__ selcnt) {
  const int t = blockIdx.x * 8 + (threadIdx.x >> 5);
  const int lane = threadIdx.x & 31;
  const float* xr = x + (size_t)t * H_DIM;
  float acc[E_NUM];
#pragma unroll
  for (int e = 0; e < E_NUM; e++) acc[e] = 0.0f;
  for (int h = lane; h < H_DIM; h += 32) {
    float xv = xr[h];
#pragma unroll
    for (int e = 0; e < E_NUM; e++) acc[e] += xv * gw[e * H_DIM + h];
  }
#pragma unroll
  for (int e = 0; e < E_NUM; e++)
    for (int off = 16; off; off >>= 1) acc[e] += __shfl_xor(acc[e], off, 32);

  if (lane == 0) {
    float mx = acc[0];
#pragma unroll
    for (int e = 1; e < E_NUM; e++) mx = fmaxf(mx, acc[e]);
    float p[E_NUM], s = 0.0f;
#pragma unroll
    for (int e = 0; e < E_NUM; e++) { p[e] = __expf(acc[e] - mx); s += p[e]; }
    float inv = __builtin_amdgcn_rcpf(s);
#pragma unroll
    for (int e = 0; e < E_NUM; e++) { p[e] *= inv; probs[t * E_NUM + e] = p[e]; }
    int i0 = 0;
#pragma unroll
    for (int e = 1; e < E_NUM; e++) if (p[e] > p[i0]) i0 = e;
    int i1 = (i0 == 0) ? 1 : 0;
#pragma unroll
    for (int e = 0; e < E_NUM; e++) if (e != i0 && p[e] > p[i1]) i1 = e;
    float winv = __builtin_amdgcn_rcpf(p[i0] + p[i1]);
#pragma unroll
    for (int e = 0; e < E_NUM; e++) padw[t * E_NUM + e] = 0.0f;
    padw[t * E_NUM + i0] = p[i0] * winv;
    padw[t * E_NUM + i1] = p[i1] * winv;
    atomicAdd(&selcnt[i0], 1);
    atomicAdd(&selcnt[i1], 1);
  }
}

// ------------------------------------------- GEMM1: hmid = silu(x w1^T)*(x w3^T)
__global__ __launch_bounds__(256) void arctic_gemm1(
    const __bf16* __restrict__ xb,   // [T, H] bf16
    const float*  __restrict__ w1,   // [F, H] f32 (expert slice)
    const float*  __restrict__ w3,   // [F, H]
    __bf16* __restrict__ hmid) {     // [T, F] bf16
  __shared__ __bf16 sA [2][BM * LDSA];
  __shared__ __bf16 sB1[2][BN * LDSB];
  __shared__ __bf16 sB3[2][BN * LDSB];

  const int m0 = blockIdx.x * BM;
  const int n0 = blockIdx.y * BN;
  const int tid  = threadIdx.x;
  const int wave = tid >> 5;
  const int lane = tid & 31;
  const int half = lane >> 4;
  const int lrow = lane & 15;
  const int wm = (wave >> 1) * 32;   // 4 waves along M
  const int wn = (wave & 1) * 32;    // 2 waves along N

  v8f acc1[2][2] = {};
  v8f acc3[2][2] = {};

  const int arow = tid >> 1, aseg = tid & 1;   // A: 2 thr/row, 16 bf16 each
  const int brow = tid >> 2, bseg = tid & 3;   // B: 4 thr/row, 8 f32 each

  const __bf16* aG  = xb + (size_t)(m0 + arow) * H_DIM + aseg * 16;
  const float*  b1G = w1 + (size_t)(n0 + brow) * H_DIM + bseg * 8;
  const float*  b3G = w3 + (size_t)(n0 + brow) * H_DIM + bseg * 8;
  const unsigned aL0 = (unsigned)(size_t)&sA[0][arow * LDSA + aseg * 16];
  const unsigned aL1 = (unsigned)(size_t)&sA[1][arow * LDSA + aseg * 16];

  v4f r1a, r1b, r3a, r3b;  // staged B regs for next tile

  auto issueA = [&](int p, int k0) {
    unsigned long long g = (unsigned long long)(const void*)(aG + k0);
    unsigned l = p ? aL1 : aL0;
    ASYNC_CP16(l, g);
    ASYNC_CP16_OFF16(l, g);
  };
  auto loadB = [&](int k0) {
    r1a = ((const v4f*)(b1G + k0))[0]; r1b = ((const v4f*)(b1G + k0))[1];
    r3a = ((const v4f*)(b3G + k0))[0]; r3b = ((const v4f*)(b3G + k0))[1];
  };
  auto storeB = [&](int p) {
    v8bf t1, t3;
#pragma unroll
    for (int i = 0; i < 4; i++) {
      t1[i] = (__bf16)r1a[i]; t1[i + 4] = (__bf16)r1b[i];
      t3[i] = (__bf16)r3a[i]; t3[i + 4] = (__bf16)r3b[i];
    }
    *(v8bf*)&sB1[p][brow * LDSB + bseg * 8] = t1;
    *(v8bf*)&sB3[p][brow * LDSB + bseg * 8] = t3;
  };

  // prologue: fill buffer 0
  issueA(0, 0);
  loadB(0);
  storeB(0);
  WAIT_ASYNC0();
  __syncthreads();

  int p = 0;
  for (int k0 = 0; k0 < H_DIM; k0 += BK) {
    const bool more = (k0 + BK) < H_DIM;
    if (more) { issueA(1 - p, k0 + BK); loadB(k0 + BK); }

    Frag a[2], b1[2], b3[2];
#pragma unroll
    for (int mi = 0; mi < 2; mi++) {
      const __bf16* ab = &sA[p][(wm + mi * 16 + lrow) * LDSA];
      a[mi].h[0] = *(const v8bf*)(ab + half * 8);        // K 0-7 / 8-15
      a[mi].h[1] = *(const v8bf*)(ab + 16 + half * 8);   // K 16-23 / 24-31
    }
#pragma unroll
    for (int ni = 0; ni < 2; ni++) {
      const __bf16* p1 = &sB1[p][(wn + ni * 16 + lrow) * LDSB + half * 8];
      const __bf16* p3 = &sB3[p][(wn + ni * 16 + lrow) * LDSB + half * 8];
      b1[ni].h[0] = ((const v8bf*)p1)[0]; b1[ni].h[1] = ((const v8bf*)p1)[2];
      b3[ni].h[0] = ((const v8bf*)p3)[0]; b3[ni].h[1] = ((const v8bf*)p3)[2];
    }
#pragma unroll
    for (int mi = 0; mi < 2; mi++)
#pragma unroll
      for (int ni = 0; ni < 2; ni++) {
        acc1[mi][ni] = __builtin_amdgcn_wmma_f32_16x16x32_bf16(
            false, a[mi].v, false, b1[ni].v, (short)0, acc1[mi][ni], false, false);
        acc3[mi][ni] = __builtin_amdgcn_wmma_f32_16x16x32_bf16(
            false, a[mi].v, false, b3[ni].v, (short)0, acc3[mi][ni], false, false);
      }

    if (more) storeB(1 - p);
    WAIT_ASYNC0();
    __syncthreads();
    p ^= 1;
  }

  // epilogue: hmid = silu(acc1) * acc3, sigmoid via hardware v_rcp_f32
#pragma unroll
  for (int mi = 0; mi < 2; mi++)
#pragma unroll
    for (int ni = 0; ni < 2; ni++) {
      const int col = n0 + wn + ni * 16 + lrow;
#pragma unroll
      for (int i = 0; i < 8; i++) {
        const int row = m0 + wm + mi * 16 + half * 8 + i;
        float z = acc1[mi][ni][i];
        float g = z * __builtin_amdgcn_rcpf(1.0f + __expf(-z));
        hmid[(size_t)row * F_DIM + col] = (__bf16)(g * acc3[mi][ni][i]);
      }
    }
}

// ------------------------------------------- GEMM2: out (+)= (hmid w2^T)*padw[:,e]
__global__ __launch_bounds__(256) void arctic_gemm2(
    const __bf16* __restrict__ hmid, // [T, F] bf16
    const float*  __restrict__ w2,   // [H, F] f32 (expert slice)
    const float*  __restrict__ padw, // [T, E]
    int expert, int first,
    float* __restrict__ out) {       // [T, H]
  __shared__ __bf16 sA[2][BM * LDSA];
  __shared__ __bf16 sB[2][BN * LDSB];

  const int m0 = blockIdx.x * BM;
  const int n0 = blockIdx.y * BN;
  const int tid  = threadIdx.x;
  const int wave = tid >> 5;
  const int lane = tid & 31;
  const int half = lane >> 4;
  const int lrow = lane & 15;
  const int wm = (wave >> 1) * 32;
  const int wn = (wave & 1) * 32;

  v8f acc[2][2] = {};

  const int arow = tid >> 1, aseg = tid & 1;
  const int brow = tid >> 2, bseg = tid & 3;

  const __bf16* aG = hmid + (size_t)(m0 + arow) * F_DIM + aseg * 16;
  const float*  bG = w2   + (size_t)(n0 + brow) * F_DIM + bseg * 8;
  const unsigned aL0 = (unsigned)(size_t)&sA[0][arow * LDSA + aseg * 16];
  const unsigned aL1 = (unsigned)(size_t)&sA[1][arow * LDSA + aseg * 16];

  v4f ra, rb;

  auto issueA = [&](int p, int k0) {
    unsigned long long g = (unsigned long long)(const void*)(aG + k0);
    unsigned l = p ? aL1 : aL0;
    ASYNC_CP16(l, g);
    ASYNC_CP16_OFF16(l, g);
  };
  auto loadB = [&](int k0) {
    ra = ((const v4f*)(bG + k0))[0]; rb = ((const v4f*)(bG + k0))[1];
  };
  auto storeB = [&](int p) {
    v8bf t;
#pragma unroll
    for (int i = 0; i < 4; i++) { t[i] = (__bf16)ra[i]; t[i + 4] = (__bf16)rb[i]; }
    *(v8bf*)&sB[p][brow * LDSB + bseg * 8] = t;
  };

  issueA(0, 0);
  loadB(0);
  storeB(0);
  WAIT_ASYNC0();
  __syncthreads();

  int p = 0;
  for (int k0 = 0; k0 < F_DIM; k0 += BK) {
    const bool more = (k0 + BK) < F_DIM;
    if (more) { issueA(1 - p, k0 + BK); loadB(k0 + BK); }

    Frag a[2], b[2];
#pragma unroll
    for (int mi = 0; mi < 2; mi++) {
      const __bf16* ab = &sA[p][(wm + mi * 16 + lrow) * LDSA];
      a[mi].h[0] = *(const v8bf*)(ab + half * 8);
      a[mi].h[1] = *(const v8bf*)(ab + 16 + half * 8);
    }
#pragma unroll
    for (int ni = 0; ni < 2; ni++) {
      const __bf16* pb = &sB[p][(wn + ni * 16 + lrow) * LDSB + half * 8];
      b[ni].h[0] = ((const v8bf*)pb)[0]; b[ni].h[1] = ((const v8bf*)pb)[2];
    }
#pragma unroll
    for (int mi = 0; mi < 2; mi++)
#pragma unroll
      for (int ni = 0; ni < 2; ni++)
        acc[mi][ni] = __builtin_amdgcn_wmma_f32_16x16x32_bf16(
            false, a[mi].v, false, b[ni].v, (short)0, acc[mi][ni], false, false);

    if (more) storeB(1 - p);
    WAIT_ASYNC0();
    __syncthreads();
    p ^= 1;
  }

  float sc[2][8];
#pragma unroll
  for (int mi = 0; mi < 2; mi++)
#pragma unroll
    for (int i = 0; i < 8; i++)
      sc[mi][i] = padw[(size_t)(m0 + wm + mi * 16 + half * 8 + i) * E_NUM + expert];

#pragma unroll
  for (int mi = 0; mi < 2; mi++)
#pragma unroll
    for (int ni = 0; ni < 2; ni++) {
      const int col = n0 + wn + ni * 16 + lrow;
#pragma unroll
      for (int i = 0; i < 8; i++) {
        const size_t idx = (size_t)(m0 + wm + mi * 16 + half * 8 + i) * H_DIM + col;
        float v = acc[mi][ni][i] * sc[mi][i];
        if (first) out[idx] = v;
        else       out[idx] += v;
      }
    }
}

// ---------------------------------------------------------------- aux loss
__global__ __launch_bounds__(256) void arctic_aux(
    const float* __restrict__ probs, const int* __restrict__ selcnt,
    float* __restrict__ auxout) {
  __shared__ float red[256];
  float sum[E_NUM];
#pragma unroll
  for (int e = 0; e < E_NUM; e++) sum[e] = 0.0f;
  for (int t = threadIdx.x; t < T_TOK; t += 256)
#pragma unroll
    for (int e = 0; e < E_NUM; e++) sum[e] += probs[t * E_NUM + e];

  float aux = 0.0f;
  for (int e = 0; e < E_NUM; e++) {
    red[threadIdx.x] = sum[e];
    __syncthreads();
    for (int s = 128; s; s >>= 1) {
      if (threadIdx.x < (unsigned)s) red[threadIdx.x] += red[threadIdx.x + s];
      __syncthreads();
    }
    if (threadIdx.x == 0)
      aux += ((float)selcnt[e] / (float)T_TOK) * (red[0] / (float)T_TOK);
    __syncthreads();
  }
  if (threadIdx.x == 0) auxout[0] = aux * (float)E_NUM;
}

// ---------------------------------------------------------------- launch
extern "C" void kernel_launch(void* const* d_in, const int* in_sizes, int n_in,
                              void* d_out, int out_size, void* d_ws, size_t ws_size,
                              hipStream_t stream) {
  const float* x      = (const float*)d_in[0];  // [T, H]
  const float* gate_w = (const float*)d_in[1];  // [E, H]
  const float* w1     = (const float*)d_in[2];  // [E, F, H]
  const float* w3     = (const float*)d_in[3];  // [E, F, H]
  const float* w2     = (const float*)d_in[4];  // [E, H, F]
  float* out = (float*)d_out;                   // T*H floats + 1 aux

  char* ws = (char*)d_ws;
  float*  probs  = (float*)ws;  ws += (size_t)T_TOK * E_NUM * 4;
  float*  padw   = (float*)ws;  ws += (size_t)T_TOK * E_NUM * 4;
  int*    selcnt = (int*)ws;    ws += 256;
  __bf16* xb     = (__bf16*)ws; ws += (size_t)T_TOK * H_DIM * 2;
  __bf16* hmid   = (__bf16*)ws; ws += (size_t)T_TOK * F_DIM * 2;

  hipMemsetAsync(selcnt, 0, 256, stream);

  arctic_cvt_bf16<<<(T_TOK * H_DIM) / 1024, 256, 0, stream>>>(x, xb, T_TOK * H_DIM);
  arctic_router<<<T_TOK / 8, 256, 0, stream>>>(x, gate_w, probs, padw, selcnt);

  dim3 g1(T_TOK / BM, F_DIM / BN);
  dim3 g2(T_TOK / BM, H_DIM / BN);
  for (int e = 0; e < E_NUM; e++) {
    arctic_gemm1<<<g1, 256, 0, stream>>>(
        xb, w1 + (size_t)e * F_DIM * H_DIM, w3 + (size_t)e * F_DIM * H_DIM, hmid);
    arctic_gemm2<<<g2, 256, 0, stream>>>(
        hmid, w2 + (size_t)e * H_DIM * F_DIM, padw, e, e == 0, out);
  }
  arctic_aux<<<1, 256, 0, stream>>>(probs, selcnt, out + (size_t)T_TOK * H_DIM);
}
